// NET_43550968381497
// MI455X (gfx1250) — matrix operand order repaired and checked by reading, not compiled, not run on hardware
//
#include <hip/hip_runtime.h>
#include <hip/hip_bf16.h>

// CDNA5 / gfx1250: wave32, WMMA 16x16x32 f16 -> f32 accumulate.
typedef _Float16 v16h __attribute__((ext_vector_type(16)));
typedef float    v8f  __attribute__((ext_vector_type(8)));

__device__ __forceinline__ float softplus_f(float x) {
    return (x > 30.0f) ? x : log1pf(expf(x));
}

__device__ __forceinline__ int imin(int a, int b) { return a < b ? a : b; }

// ---------------------------------------------------------------------------
// Kernel 0: zero the segment-sum workspace (harness poisons d_ws with 0xAA).
// ---------------------------------------------------------------------------
__global__ void zero_ws_kernel(float* __restrict__ p, int n) {
    int i = blockIdx.x * blockDim.x + threadIdx.x;
    if (i < n) p[i] = 0.0f;
}

// ---------------------------------------------------------------------------
// Kernel 1: per-atom featurization. One wave per atom, 8 waves / block.
//   stage 1: conva (1->3ch K=20) VALU fp32, weights in LDS
//   stage 2: conv1 (1->3ch K=20) + relu + maxpool2 -> one[3][127] (stride 128)
//   stage 3: conv2 (3->6ch K=20) as WMMA: M = 7x16 positions, K = 60 pad 64,
//            N = 6 of 16; bias+relu+maxpool2; segment-sum via global atomics.
// A-fragment gather is fully branchless: precomputed per-lane LDS offsets,
// invalid elements redirected (v_cndmask + v_min) to a zeroed pad slot.
// ---------------------------------------------------------------------------
__global__ __launch_bounds__(256) void featurize_kernel(
    const float* __restrict__ x_onehot, const float* __restrict__ x_env,
    const int*   __restrict__ cidx,
    const float* __restrict__ wa, const float* __restrict__ ba,
    const float* __restrict__ w1, const float* __restrict__ b1,
    const float* __restrict__ w2, const float* __restrict__ b2,
    float* __restrict__ sums, float* __restrict__ counts, int m0)
{
    __shared__ float s_wa[60], s_ba[3], s_w1[60], s_b1[3];
    __shared__ float s_x  [8][96];    // one-hot input (92)
    __shared__ float s_tot[8][280];   // concat(conva_out(219), env(55)) = 274
    __shared__ float s_one[8][400];   // conv1 pooled out, stride 128/ch + pad

    const int tid = threadIdx.x;
    const int wv  = tid >> 5;
    const int ln  = tid & 31;
    const int n   = ln & 15;          // WMMA N (cout) / M-row selector
    const bool hi = (ln >= 16);
    const int h8  = hi ? 8 : 0;

    for (int i = tid; i < 60; i += 256) { s_wa[i] = wa[i]; s_w1[i] = w1[i]; }
    if (tid < 3) { s_ba[tid] = ba[tid]; s_b1[tid] = b1[tid]; }
    if (ln < 16) s_one[wv][384 + ln] = 0.0f;   // zero pad slots (incl. 384)
    __syncthreads();

    // conv2 weights as WMMA B fragments (constant across atoms).
    // B layout (32x16 f16): N = lane%16; halves 0-7 -> K 0..7, 8-15 -> K 16..23, +8*hi.
    v16h blo, bhi;
    #pragma unroll
    for (int j = 0; j < 16; ++j) {
        const int kl = (j < 8 ? j : j + 8) + h8;
        float v0 = 0.0f, v1 = 0.0f;
        if (n < 6) {
            const int k0 = kl, k1 = kl + 32;           // K = cin*20 + k
            if (k0 < 60) v0 = w2[(n * 3 + k0 / 20) * 20 + (k0 % 20)];
            if (k1 < 60) v1 = w2[(n * 3 + k1 / 20) * 20 + (k1 % 20)];
        }
        blo[j] = (_Float16)v0;
        bhi[j] = (_Float16)v1;
    }
    const float b2n = (n < 6) ? b2[n] : 0.0f;

    // Per-lane constant LDS offsets for the A-fragment gather.
    // A layout (16x32 f16): halves 0-7 -> K 0..7, halves 8-15 -> K 16..23;
    // lanes 16-31 add +8. offset(kappa) = (kappa/20)*128 + kappa%20.
    int offA1[16], offA2[16];
    #pragma unroll
    for (int j = 0; j < 16; ++j) {
        const int kl = (j < 8 ? j : j + 8) + h8;
        const int kA = kl, kB = kl + 32;
        offA1[j] = (kA < 60) ? ((kA / 20) * 128 + (kA % 20)) : 1000;
        offA2[j] = (kB < 60) ? ((kB / 20) * 128 + (kB % 20)) : 1000;
    }

    const int gw     = blockIdx.x * 8 + wv;
    const int nwaves = gridDim.x * 8;
    const int iters  = (m0 + nwaves - 1) / nwaves;
    float* const s_flat = &s_one[wv][0];

    for (int it = 0; it < iters; ++it) {
        const int a = it * nwaves + gw;
        const bool active = (a < m0);

        if (active) {
            const float* xp = x_onehot + (size_t)a * 92;
            for (int i = ln; i < 92; i += 32) s_x[wv][i] = xp[i];
            const float* ep = x_env + (size_t)a * 55;
            for (int i = ln; i < 55; i += 32) s_tot[wv][219 + i] = ep[i];
            if (a + nwaves < m0)   // prefetch next atom's one-hot row
                __builtin_prefetch(x_onehot + (size_t)(a + nwaves) * 92, 0, 1);
        }
        __syncthreads();

        if (active) {
            // stage 1: conva + relu -> total[c*73 + p]
            for (int p = ln; p < 73; p += 32) {
                #pragma unroll
                for (int c = 0; c < 3; ++c) {
                    float s = s_ba[c];
                    #pragma unroll
                    for (int k = 0; k < 20; ++k)
                        s = fmaf(s_wa[c * 20 + k], s_x[wv][p + k], s);
                    s_tot[wv][c * 73 + p] = fmaxf(s, 0.0f);
                }
            }
        }
        __syncthreads();

        if (active) {
            // stage 2: conv1 + relu + maxpool2 -> one[c*128 + q]
            for (int q = ln; q < 127; q += 32) {
                #pragma unroll
                for (int c = 0; c < 3; ++c) {
                    float m = 0.0f;   // relu outputs >= 0
                    #pragma unroll
                    for (int t = 0; t < 2; ++t) {
                        const int p = 2 * q + t;
                        float s = s_b1[c];
                        #pragma unroll
                        for (int k = 0; k < 20; ++k)
                            s = fmaf(s_w1[c * 20 + k], s_tot[wv][p + k], s);
                        m = fmaxf(m, s);
                    }
                    s_one[wv][c * 128 + q] = m;
                }
            }
        }
        __syncthreads();

        if (active) {
            const int crys = cidx[a];
            if (ln == 0) atomicAdd(&counts[crys], 1.0f);
            float* srow = sums + (size_t)crys * 324;

            #pragma unroll 1
            for (int t = 0; t < 7; ++t) {
                const int q  = t * 16 + n;      // A row (both halves same row)
                const bool qa = (q < 108);
                v16h alo, ahi;
                #pragma unroll
                for (int j = 0; j < 16; ++j) {
                    // branchless: invalid -> pad slot 384 (known zero)
                    int i1 = qa ? (q + offA1[j]) : 1000;
                    int i2 = qa ? (q + offA2[j]) : 1000;
                    alo[j] = (_Float16)s_flat[imin(i1, 384)];
                    ahi[j] = (_Float16)s_flat[imin(i2, 384)];
                }
                v8f acc = {};
                acc = __builtin_amdgcn_wmma_f32_16x16x32_f16(
                        false, alo, false, blo, (short)0, acc, false, false);
                acc = __builtin_amdgcn_wmma_f32_16x16x32_f16(
                        false, ahi, false, bhi, (short)0, acc, false, false);

                // C layout: M = vgpr + 8*hi, N = lane%16; maxpool pairs are
                // adjacent accumulator entries of the same lane.
                if (n < 6) {
                    #pragma unroll
                    for (int j2 = 0; j2 < 4; ++j2) {
                        const int s = t * 8 + (hi ? 4 : 0) + j2;
                        if (s < 54) {
                            const float v0 = fmaxf(acc[2 * j2]     + b2n, 0.0f);
                            const float v1 = fmaxf(acc[2 * j2 + 1] + b2n, 0.0f);
                            atomicAdd(&srow[n * 54 + s], fmaxf(v0, v1));
                        }
                    }
                }
            }
        }
        __syncthreads();
    }
}

// ---------------------------------------------------------------------------
// Kernel 2: pooled = relu(sums/count); h = softplus(pooled @ lin_w^T + lin_b);
//           out = h @ lin1_w^T + lin1_b.
// GEMM M=ncry, N=32 (2 tiles), K=324 (10 unmasked kb steps of 32 via float4
// loads + 1 clamped/masked tail step). One wave per 16 rows.
// ---------------------------------------------------------------------------
__global__ __launch_bounds__(128) void head_kernel(
    const float* __restrict__ sums, const float* __restrict__ counts,
    const float* __restrict__ lin_w, const float* __restrict__ lin_b,
    const float* __restrict__ lin1_w, const float* __restrict__ lin1_b,
    float* __restrict__ out, int ncry)
{
    const int tid = threadIdx.x;
    const int wv  = tid >> 5;
    const int ln  = tid & 31;
    const int n   = ln & 15;
    const bool hi = (ln >= 16);
    const int h8  = hi ? 8 : 0;
    const int h16 = hi ? 16 : 0;

    const int r0 = blockIdx.x * 64 + wv * 16;
    int row = imin(r0 + n, ncry - 1);               // clamp loads
    const float invc = 1.0f / fmaxf(counts[row], 1.0f);
    const float* arow = sums  + (size_t)row * 324;  // 16B aligned (324%4==0)
    const float* brow0 = lin_w + (size_t)n * 324;
    const float* brow1 = lin_w + (size_t)(n + 16) * 324;

    v8f acc0 = {}, acc1 = {};

    #pragma unroll 1
    for (int kb = 0; kb < 320; kb += 32) {          // no masking needed (<324)
        // A fragment: halves 0..15 <-> K kb+h16+0..15 (contiguous)
        const float4 a0 = *(const float4*)(arow + kb + h16);
        const float4 a1 = *(const float4*)(arow + kb + h16 + 4);
        const float4 a2 = *(const float4*)(arow + kb + h16 + 8);
        const float4 a3 = *(const float4*)(arow + kb + h16 + 12);
        // B fragments: halves 0..15 <-> K kb+h16+0..15 (contiguous)
        const float4 b0 = *(const float4*)(brow0 + kb + h16);
        const float4 b1 = *(const float4*)(brow0 + kb + h16 + 4);
        const float4 b2 = *(const float4*)(brow0 + kb + h16 + 8);
        const float4 b3 = *(const float4*)(brow0 + kb + h16 + 12);
        const float4 c0 = *(const float4*)(brow1 + kb + h16);
        const float4 c1 = *(const float4*)(brow1 + kb + h16 + 4);
        const float4 c2 = *(const float4*)(brow1 + kb + h16 + 8);
        const float4 c3 = *(const float4*)(brow1 + kb + h16 + 12);

        v16h a, bA, bB;
        const float av[16] = { a0.x, a0.y, a0.z, a0.w, a1.x, a1.y, a1.z, a1.w,
                               a2.x, a2.y, a2.z, a2.w, a3.x, a3.y, a3.z, a3.w };
        const float bv[16] = { b0.x, b0.y, b0.z, b0.w, b1.x, b1.y, b1.z, b1.w,
                               b2.x, b2.y, b2.z, b2.w, b3.x, b3.y, b3.z, b3.w };
        const float cv[16] = { c0.x, c0.y, c0.z, c0.w, c1.x, c1.y, c1.z, c1.w,
                               c2.x, c2.y, c2.z, c2.w, c3.x, c3.y, c3.z, c3.w };
        #pragma unroll
        for (int j = 0; j < 16; ++j) {
            a [j] = (_Float16)fmaxf(av[j] * invc, 0.0f);
            bA[j] = (_Float16)bv[j];
            bB[j] = (_Float16)cv[j];
        }
        acc0 = __builtin_amdgcn_wmma_f32_16x16x32_f16(
                 false, a, false, bA, (short)0, acc0, false, false);
        acc1 = __builtin_amdgcn_wmma_f32_16x16x32_f16(
                 false, a, false, bB, (short)0, acc1, false, false);
    }

    {   // tail step kb = 320: clamp indices (no OOB) + mask values to zero
        v16h a, bA, bB;
        #pragma unroll
        for (int j = 0; j < 16; ++j) {
            const int ka = 320 + j + h16;
            const int kk = 320 + j + h16;
            const float av = arow [imin(ka, 323)];
            const float b0 = brow0[imin(kk, 323)];
            const float b1 = brow1[imin(kk, 323)];
            a [j] = (_Float16)((ka < 324) ? fmaxf(av * invc, 0.0f) : 0.0f);
            bA[j] = (_Float16)((kk < 324) ? b0 : 0.0f);
            bB[j] = (_Float16)((kk < 324) ? b1 : 0.0f);
        }
        acc0 = __builtin_amdgcn_wmma_f32_16x16x32_f16(
                 false, a, false, bA, (short)0, acc0, false, false);
        acc1 = __builtin_amdgcn_wmma_f32_16x16x32_f16(
                 false, a, false, bB, (short)0, acc1, false, false);
    }

    const float bn0 = lin_b[n], bn1 = lin_b[n + 16];
    const float w0  = lin1_w[n], w1 = lin1_w[n + 16];
    const float ob  = lin1_b[0];

    #pragma unroll
    for (int v = 0; v < 8; ++v) {
        float t = softplus_f(acc0[v] + bn0) * w0 + softplus_f(acc1[v] + bn1) * w1;
        // reduce the 32 output columns within each 16-lane group (wave32)
        t += __shfl_xor(t, 1, 32);
        t += __shfl_xor(t, 2, 32);
        t += __shfl_xor(t, 4, 32);
        t += __shfl_xor(t, 8, 32);
        const int orow = r0 + v + (hi ? 8 : 0);
        if (n == 0 && orow < ncry) out[orow] = t + ob;
    }
}

// ---------------------------------------------------------------------------
extern "C" void kernel_launch(void* const* d_in, const int* in_sizes, int n_in,
                              void* d_out, int out_size, void* d_ws, size_t ws_size,
                              hipStream_t stream) {
    const float* x   = (const float*)d_in[0];
    const float* env = (const float*)d_in[1];
    const int*   ci  = (const int*)  d_in[2];
    // d_in[3] = num_segments (device scalar; n_crystals known from out_size)
    const float* wa  = (const float*)d_in[4];
    const float* ba  = (const float*)d_in[5];
    const float* w1  = (const float*)d_in[6];
    const float* b1  = (const float*)d_in[7];
    const float* w2  = (const float*)d_in[8];
    const float* b2  = (const float*)d_in[9];
    const float* lw  = (const float*)d_in[10];
    const float* lb  = (const float*)d_in[11];
    const float* l1w = (const float*)d_in[12];
    const float* l1b = (const float*)d_in[13];

    const int m0   = in_sizes[0] / 92;
    const int ncry = (out_size > 0) ? out_size : 4096;

    float* sums   = (float*)d_ws;                 // [ncry * 324]
    float* counts = sums + (size_t)ncry * 324;    // [ncry]

    const int nzero = ncry * 325;
    zero_ws_kernel<<<(nzero + 255) / 256, 256, 0, stream>>>(sums, nzero);

    featurize_kernel<<<512, 256, 0, stream>>>(x, env, ci, wa, ba, w1, b1,
                                              w2, b2, sums, counts, m0);

    head_kernel<<<(ncry + 63) / 64, 128, 0, stream>>>(sums, counts, lw, lb,
                                                      l1w, l1b, (float*)d_out, ncry);
}
